// MultiHeadAttention_63410897158394
// MI455X (gfx1250) — matrix operand order repaired and checked by reading, not compiled, not run on hardware
//
#include <hip/hip_runtime.h>

namespace {
constexpr int B = 8, S = 1024, D = 768, H = 12, DH = 64;
constexpr int BH = B * H; // 96
}

typedef __attribute__((ext_vector_type(16))) _Float16 v16h;
typedef __attribute__((ext_vector_type(8)))  float    v8f;
typedef __attribute__((ext_vector_type(4)))  int      v4i_t;

__device__ __forceinline__ v8f wmma16(v16h a, v16h b, v8f c) {
  // D = A(16x32 f16) * B(32x16 f16) + C(16x16 f32)
  return __builtin_amdgcn_wmma_f32_16x16x32_f16(false, a, false, b, (short)0, c,
                                                false, false);
}

// A-fragment loader: halves[0:16] = row[2*base .. 2*base+15] (contiguous k)
__device__ __forceinline__ v16h load_a_frag(const _Float16* row, int base) {
  union { v16h v; uint4 q[2]; } u;
  u.q[0] = *(const uint4*)(row + 2 * base);
  u.q[1] = *(const uint4*)(row + 2 * base + 8);
  return u.v;
}

// ---- CDNA5 async global->LDS copy (ASYNCcnt path), with portable fallback ----
#if defined(__HIP_DEVICE_COMPILE__)
#if __has_builtin(__builtin_amdgcn_global_load_async_to_lds_b128)
#define HAVE_ASYNC_LDS 1
#endif
#endif

#if defined(HAVE_ASYNC_LDS)
__device__ __forceinline__ void cp16_g2l(_Float16* lds, const _Float16* g) {
  __builtin_amdgcn_global_load_async_to_lds_b128((v4i_t*)g, (v4i_t*)lds, 0, 0);
}
#if __has_builtin(__builtin_amdgcn_s_wait_asynccnt)
#define ASYNC_WAIT() __builtin_amdgcn_s_wait_asynccnt(0)
#else
#define ASYNC_WAIT() asm volatile("s_wait_asynccnt 0x0" ::: "memory")
#endif
#else
__device__ __forceinline__ void cp16_g2l(_Float16* lds, const _Float16* g) {
  *(uint4*)lds = *(const uint4*)g; // global_load_b128 + ds_store_b128
}
#define ASYNC_WAIT() do {} while (0)
#endif

// ----------------- conversion / transpose kernels -----------------
__global__ void k_cvt_f32_f16(const float* __restrict__ src,
                              _Float16* __restrict__ dst, int n) {
  for (int i = blockIdx.x * blockDim.x + threadIdx.x; i < n;
       i += gridDim.x * blockDim.x)
    dst[i] = (_Float16)src[i];
}

// W: [H][D][DH] f32 -> Wt: [H][DH][D] f16 (output-major, K-contiguous)
__global__ void k_cvt_w_qkv(const float* __restrict__ w,
                            _Float16* __restrict__ wt) {
  const int n = H * D * DH;
  for (int i = blockIdx.x * blockDim.x + threadIdx.x; i < n;
       i += gridDim.x * blockDim.x) {
    const int e = i % DH;
    const int d = (i / DH) % D;
    const int h = i / (DH * D);
    wt[((size_t)h * DH + e) * D + d] = (_Float16)w[i];
  }
}

// Wo: [D(in)][D(out)] f32 -> Wot: [out][in] f16
__global__ void k_cvt_wo(const float* __restrict__ w, _Float16* __restrict__ wt) {
  const int n = D * D;
  for (int i = blockIdx.x * blockDim.x + threadIdx.x; i < n;
       i += gridDim.x * blockDim.x) {
    const int e = i % D;
    const int d = i / D;
    wt[(size_t)e * D + d] = (_Float16)w[i];
  }
}

// ----------------- fused QKV projection -----------------
// grid (16, BH), block 128 (4 waves). Wave: 16 rows x 64 cols of q,k,v for one (b,h).
__global__ void __launch_bounds__(128)
k_qkv_proj(const _Float16* __restrict__ xh,
           const _Float16* __restrict__ wqt, const _Float16* __restrict__ wkt,
           const _Float16* __restrict__ wvt,
           const float* __restrict__ bq, const float* __restrict__ bk,
           const float* __restrict__ bv,
           _Float16* __restrict__ qh, _Float16* __restrict__ kh,
           _Float16* __restrict__ vt) {
  const int wave = threadIdx.x >> 5;
  const int lane = threadIdx.x & 31;
  const int l15 = lane & 15, half = lane >> 4;
  const int abase = half * 8, kbase = half * 16, mrow = half * 8;
  const int stile = blockIdx.x * 4 + wave; // 0..63
  const int bh = blockIdx.y;               // 0..95
  const int b = bh / H, h = bh % H;
  const int s0 = stile * 16;

  const _Float16* xrow = xh + ((size_t)b * S + s0 + l15) * D;
  const _Float16* wqh = wqt + (size_t)h * DH * D;
  const _Float16* wkh = wkt + (size_t)h * DH * D;
  const _Float16* wvh = wvt + (size_t)h * DH * D;

  v8f aq[4] = {}, ak[4] = {}, av[4] = {};

  for (int k0 = 0; k0 < D; k0 += 32) {
    const v16h a = load_a_frag(xrow + k0, abase);
#pragma unroll
    for (int f = 0; f < 4; ++f) {
      const size_t wo = (size_t)(f * 16 + l15) * D + k0 + kbase;
      const v16h bqf = *(const v16h*)(wqh + wo);
      const v16h bkf = *(const v16h*)(wkh + wo);
      const v16h bvf = *(const v16h*)(wvh + wo);
      aq[f] = wmma16(a, bqf, aq[f]);
      ak[f] = wmma16(a, bkf, ak[f]);
      av[f] = wmma16(a, bvf, av[f]);
    }
  }

  _Float16* qrow = qh + ((size_t)bh * S + s0) * DH;
  _Float16* krow = kh + ((size_t)bh * S + s0) * DH;
#pragma unroll
  for (int f = 0; f < 4; ++f) {
    const int e = f * 16 + l15;
    const float bqv = bq[h * DH + e], bkv = bk[h * DH + e], bvv = bv[h * DH + e];
#pragma unroll
    for (int i = 0; i < 8; ++i) {
      qrow[(size_t)(mrow + i) * DH + e] = (_Float16)(aq[f][i] + bqv);
      krow[(size_t)(mrow + i) * DH + e] = (_Float16)(ak[f][i] + bkv);
    }
    // V stored transposed [b,h,e,s] so the PV GEMM B-fragment is K-contiguous.
    union { uint4 u; _Float16 hx[8]; } pv;
#pragma unroll
    for (int i = 0; i < 8; ++i) pv.hx[i] = (_Float16)(av[f][i] + bvv);
    *(uint4*)(vt + ((size_t)bh * DH + e) * S + s0 + mrow) = pv.u;
  }
}

// ----------------- flash-style attention with LDS K/V staging -----------------
// grid (16, BH), block 128 (4 waves, same (b,h), different 16-row q tiles).
// K tile [32 x 64] and V^T tile [64 x 32] staged in LDS once per block,
// double-buffered via async global->LDS copies; 8 WMMA per 32-key step.
__global__ void __launch_bounds__(128)
k_attn(const _Float16* __restrict__ qh, const _Float16* __restrict__ kh,
       const _Float16* __restrict__ vt, _Float16* __restrict__ ctx) {
  __shared__ _Float16 ktile[2][32 * DH]; // [key][feat], linear copy of K rows
  __shared__ _Float16 vtile[2][DH * 32]; // [feat][key], rows of V^T
  __shared__ _Float16 pbuf[4][16 * 32];  // per-wave P tile (C-layout -> A-layout)
  const int tid = threadIdx.x;
  const int wave = tid >> 5;
  const int lane = tid & 31;
  const int l15 = lane & 15, half = lane >> 4;
  const int abase = half * 8, kbase = half * 16, mrow = half * 8;
  const int qt = blockIdx.x * 4 + wave;
  const int bh = blockIdx.y;
  const int b = bh / H, h = bh % H;
  const int s0 = qt * 16;

  const _Float16* kb = kh + (size_t)bh * S * DH;
  const _Float16* vb = vt + (size_t)bh * DH * S;
  const _Float16* qrow = qh + ((size_t)bh * S + s0 + l15) * DH;

  const v16h aq0 = load_a_frag(qrow, abase);      // features 0..31
  const v16h aq1 = load_a_frag(qrow + 32, abase); // features 32..63

  float mrun[8], lrun[8];
  v8f acc[4] = {};
#pragma unroll
  for (int i = 0; i < 8; ++i) { mrun[i] = -3.0e38f; lrun[i] = 0.f; }

  _Float16* myp = pbuf[wave];
  const float scale = 0.125f; // 1/sqrt(64)

  // Stage K/V tiles for key block starting at t into buffer `buf`.
  // 256 16-byte chunks per tile, 128 threads -> 2 chunks each per tile.
  auto stage = [&](int buf, int t) {
#pragma unroll
    for (int r = 0; r < 2; ++r) {
      const int c = tid + r * 128; // 0..255
      cp16_g2l(&ktile[buf][c * 8], kb + (size_t)t * DH + c * 8);
      const int f = c >> 2, o = (c & 3) * 8;
      cp16_g2l(&vtile[buf][c * 8], vb + (size_t)f * S + t + o);
    }
  };

  stage(0, 0);

  for (int t = 0; t < S; t += 32) {
    const int cur = (t >> 5) & 1;
    ASYNC_WAIT();     // this wave's async copies for buffer `cur` complete
    __syncthreads();  // all waves' staging visible
    if (t + 32 < S) stage(cur ^ 1, t + 32); // prefetch next tile during compute

    const _Float16* kt = ktile[cur];
    const _Float16* vl = vtile[cur];

    // S tile [16 q x 32 keys] = Q @ K^T  (B-fragments from LDS)
    v8f c0 = {}, c1 = {};
    c0 = wmma16(aq0, *(const v16h*)(kt + (size_t)l15 * DH + kbase), c0);
    c0 = wmma16(aq1, *(const v16h*)(kt + (size_t)l15 * DH + 32 + kbase), c0);
    c1 = wmma16(aq0, *(const v16h*)(kt + (size_t)(16 + l15) * DH + kbase), c1);
    c1 = wmma16(aq1, *(const v16h*)(kt + (size_t)(16 + l15) * DH + 32 + kbase), c1);

    float alpha[8];
#pragma unroll
    for (int i = 0; i < 8; ++i) {
      const float x0 = c0[i] * scale, x1 = c1[i] * scale;
      float mx = fmaxf(x0, x1);
#pragma unroll
      for (int m = 1; m < 16; m <<= 1) mx = fmaxf(mx, __shfl_xor(mx, m, 32));
      const float mnew = fmaxf(mrun[i], mx);
      const float al = __expf(mrun[i] - mnew);
      const float p0 = __expf(x0 - mnew);
      const float p1 = __expf(x1 - mnew);
      float rs = p0 + p1;
#pragma unroll
      for (int m = 1; m < 16; m <<= 1) rs += __shfl_xor(rs, m, 32);
      lrun[i] = lrun[i] * al + rs;
      mrun[i] = mnew;
      alpha[i] = al;
      const int row = mrow + i;
      myp[row * 32 + l15] = (_Float16)p0;
      myp[row * 32 + 16 + l15] = (_Float16)p1;
    }
#pragma unroll
    for (int f = 0; f < 4; ++f)
#pragma unroll
      for (int i = 0; i < 8; ++i) acc[f][i] *= alpha[i];

    __syncthreads();
    const v16h ap = load_a_frag(myp + l15 * 32, abase); // P as A-fragment
    __syncthreads();

#pragma unroll
    for (int f = 0; f < 4; ++f) { // acc += P @ V, V^T rows from LDS
      const _Float16* vr = vl + (size_t)(f * 16 + l15) * 32 + kbase;
      acc[f] = wmma16(ap, *(const v16h*)vr, acc[f]);
    }
  }

  _Float16* crow = ctx + ((size_t)b * S + s0) * D + h * DH; // head-concat layout
#pragma unroll
  for (int f = 0; f < 4; ++f)
#pragma unroll
    for (int i = 0; i < 8; ++i)
      crow[(size_t)(mrow + i) * D + f * 16 + l15] = (_Float16)(acc[f][i] / lrun[i]);
}

// ----------------- output projection -----------------
// grid (128, 12), block 128. Wave: 16 rows x 64 cols of out.
__global__ void __launch_bounds__(128)
k_out_proj(const _Float16* __restrict__ ctxh, const _Float16* __restrict__ wot,
           const float* __restrict__ bo, float* __restrict__ out) {
  const int wave = threadIdx.x >> 5, lane = threadIdx.x & 31;
  const int l15 = lane & 15, half = lane >> 4;
  const int abase = half * 8, kbase = half * 16, mrow = half * 8;
  const int rt = blockIdx.x * 4 + wave; // 0..511 over B*S/16 rows
  const int e0 = blockIdx.y * 64;
  const int s0 = rt * 16;

  const _Float16* arow = ctxh + (size_t)(s0 + l15) * D;
  v8f acc[4] = {};
  for (int k0 = 0; k0 < D; k0 += 32) {
    const v16h a = load_a_frag(arow + k0, abase);
#pragma unroll
    for (int f = 0; f < 4; ++f) {
      const _Float16* br = wot + (size_t)(e0 + f * 16 + l15) * D + k0 + kbase;
      acc[f] = wmma16(a, *(const v16h*)br, acc[f]);
    }
  }
#pragma unroll
  for (int f = 0; f < 4; ++f) {
    const int e = e0 + f * 16 + l15;
    const float bias = bo[e];
    float* orow = out + (size_t)s0 * D + e;
#pragma unroll
    for (int i = 0; i < 8; ++i) orow[(size_t)(mrow + i) * D] = acc[f][i] + bias;
  }
}

extern "C" void kernel_launch(void* const* d_in, const int* in_sizes, int n_in,
                              void* d_out, int out_size, void* d_ws, size_t ws_size,
                              hipStream_t stream) {
  const float* x  = (const float*)d_in[0];
  const float* Wq = (const float*)d_in[1];
  const float* Wk = (const float*)d_in[2];
  const float* Wv = (const float*)d_in[3];
  const float* bq = (const float*)d_in[4];
  const float* bk = (const float*)d_in[5];
  const float* bv = (const float*)d_in[6];
  const float* Wo = (const float*)d_in[7];
  const float* bo = (const float*)d_in[8];
  float* out = (float*)d_out;

  char* ws = (char*)d_ws;
  size_t off = 0;
  auto alloc = [&](size_t bytes) -> char* {
    char* p = ws + off;
    off += (bytes + 255) & ~(size_t)255;
    return p;
  };
  _Float16* xh  = (_Float16*)alloc((size_t)B * S * D * 2);   // x in f16
  _Float16* wqt = (_Float16*)alloc((size_t)H * DH * D * 2);  // Wq^T f16
  _Float16* wkt = (_Float16*)alloc((size_t)H * DH * D * 2);
  _Float16* wvt = (_Float16*)alloc((size_t)H * DH * D * 2);
  _Float16* wot = (_Float16*)alloc((size_t)D * D * 2);       // Wo^T f16
  _Float16* qh  = (_Float16*)alloc((size_t)BH * S * DH * 2); // [b,h,s,e]
  _Float16* kh  = (_Float16*)alloc((size_t)BH * S * DH * 2); // [b,h,s,e]
  _Float16* vtb = (_Float16*)alloc((size_t)BH * DH * S * 2); // [b,h,e,s]
  _Float16* cth = (_Float16*)alloc((size_t)B * S * D * 2);   // ctx f16

  k_cvt_f32_f16<<<2048, 256, 0, stream>>>(x, xh, B * S * D);
  k_cvt_w_qkv<<<576, 256, 0, stream>>>(Wq, wqt);
  k_cvt_w_qkv<<<576, 256, 0, stream>>>(Wk, wkt);
  k_cvt_w_qkv<<<576, 256, 0, stream>>>(Wv, wvt);
  k_cvt_wo<<<576, 256, 0, stream>>>(Wo, wot);

  k_qkv_proj<<<dim3(16, BH), 128, 0, stream>>>(xh, wqt, wkt, wvt, bq, bk, bv,
                                               qh, kh, vtb);
  k_attn<<<dim3(16, BH), 128, 0, stream>>>(qh, kh, vtb, cth);
  k_out_proj<<<dim3(128, 12), 128, 0, stream>>>(cth, wot, bo, out);
}
